// LSTM_31439160607432
// MI455X (gfx1250) — matrix operand order repaired and checked by reading, not compiled, not run on hardware
//
#include <hip/hip_runtime.h>
#include <hip/hip_fp16.h>

// ---- problem constants (B=16, S=8192, H=256, L=2) ----
#define H       256
#define LAYERS  2
#define ROWS    (16 * 8192)                 // B*S = 131072 flattened rows
#define OUTCOPY ((size_t)ROWS * H * LAYERS) // 67,108,864 floats per hidden copy
#define CELLS   (LAYERS * 16 * H * LAYERS)  // 16,384 zero floats at the tail

typedef __attribute__((ext_vector_type(16))) _Float16 v16h;
typedef __attribute__((ext_vector_type(8)))  _Float16 v8h;
typedef __attribute__((ext_vector_type(8)))  float    v8f;
typedef __attribute__((ext_vector_type(4)))  float    v4f;   // native vector for NT loads

// CDNA5 has a hardware V_TANH_F32 transcendental; use it if the toolchain
// exposes a builtin for it, otherwise fall back to libm.
__device__ __forceinline__ float fast_tanh(float x) {
#if __has_builtin(__builtin_amdgcn_tanhf)
    return __builtin_amdgcn_tanhf(x);
#elif __has_builtin(__builtin_amdgcn_tanh_f32)
    return __builtin_amdgcn_tanh_f32(x);
#else
    return tanhf(x);
#endif
}
// sigmoid(x) = 0.5*tanh(x/2) + 0.5  (one trans op instead of exp+div)
__device__ __forceinline__ float fast_sigmoid(float x) {
    return __builtin_fmaf(0.5f, fast_tanh(0.5f * x), 0.5f);
}

// ---------------------------------------------------------------------------
// Prep: convert the 3 live gate weights (Wi,Wo,Wg; Wf is dead) to f16 and
// transpose to Wt[g][layer][n][k] so WMMA B-fragments are K-contiguous.
// idx = ((g*2 + l) << 16) | (n << 8) | k
// ---------------------------------------------------------------------------
__global__ void prep_wt(const float* __restrict__ Wi,
                        const float* __restrict__ Wo,
                        const float* __restrict__ Wg,
                        _Float16* __restrict__ Wt) {
    int idx = blockIdx.x * blockDim.x + threadIdx.x;   // 3*2*256*256 = 393216
    int k = idx & (H - 1);
    int n = (idx >> 8) & (H - 1);
    int l = (idx >> 16) & 1;
    int g = idx >> 17;
    const float* W = (g == 0) ? Wi : (g == 1) ? Wo : Wg;
    Wt[idx] = (_Float16)W[((size_t)l * H + k) * H + n];   // W[l][k][n] -> Wt[..][n][k]
}

// output_cell block is all zeros in the reference; d_out is poisoned, so write it.
__global__ void zero_cells(float* __restrict__ out) {
    int i = blockIdx.x * blockDim.x + threadIdx.x;     // 16384 threads
    __builtin_nontemporal_store(0.0f, &out[2 * OUTCOPY + i]);
}

// ---------------------------------------------------------------------------
// Fused 6-GEMM + activation kernel.
// Block = 512 threads = 16 waves (wave32). Block tile: M=64, N=256 (full),
// both layers looped over one LDS-staged x tile (x read exactly once).
// Wave (wm in 0..3, wn in 0..3): M-strip of 16 rows, N-strip of 64 cols,
// all 3 gates accumulated in-register -> fused epilogue, dual NT store.
// ---------------------------------------------------------------------------
__launch_bounds__(512)
__global__ void lstm_gemm(const float*    __restrict__ x,
                          const _Float16* __restrict__ Wt,
                          const float*    __restrict__ bi,
                          const float*    __restrict__ bo,
                          const float*    __restrict__ bg,
                          float*          __restrict__ out) {
    // 64 rows x (256+16) halves; stride 544B keeps v8h chunks 16B-aligned
    // and spreads LDS banks across rows.
    __shared__ __align__(32) _Float16 As[64][H + 16];

    const int tid  = threadIdx.x;
    const int lane = tid & 31;
    const int wave = tid >> 5;          // 0..15
    const int wm   = wave & 3;          // M sub-tile (16 rows each)
    const int wn   = wave >> 2;         // N sub-tile (64 cols each)
    const int m0   = blockIdx.x * 64;

    // ---- stage x tile (64 x 256 fp32 -> f16) into LDS, once per block ----
    // x is a read-once stream: non-temporal so the f16 weights stay in cache.
    #pragma unroll
    for (int c = 0; c < 8; ++c) {
        int slot = c * 512 + tid;               // 4096 float4 slots
        int row  = slot >> 6;                   // 64 float4 per row
        int col4 = slot & 63;
        const v4f v = __builtin_nontemporal_load(
            (const v4f*)(x + (size_t)(m0 + row) * H) + col4);
        _Float16* dst = &As[row][col4 * 4];
        dst[0] = (_Float16)v.x;  dst[1] = (_Float16)v.y;
        dst[2] = (_Float16)v.z;  dst[3] = (_Float16)v.w;
    }
    __syncthreads();

    const int halfSel = lane >> 4;      // 0: lanes 0-15, 1: lanes 16-31
    const int lidx    = lane & 15;
    const int arow    = wm * 16 + lidx; // A-matrix row owned by this lane
    const int ncol    = wn * 64 + lidx; // base N column (per 16-wide tile)

    for (int layer = 0; layer < LAYERS; ++layer) {
        v8f acc[3][4];
        #pragma unroll
        for (int g = 0; g < 3; ++g)
            #pragma unroll
            for (int t = 0; t < 4; ++t)
                acc[g][t] = (v8f){0.f, 0.f, 0.f, 0.f, 0.f, 0.f, 0.f, 0.f};

        // ---- K loop: 256 = 8 steps of 32 ----
        #pragma unroll
        for (int ks = 0; ks < 8; ++ks) {
            // A fragment: lane holds two 8-half chunks,
            // K = ks*32 + 16*halfSel + {0..7} and +8 (16 contiguous K, matches B).
            union { v16h v; v8h h[2]; } a;
            const int ka = ks * 32 + halfSel * 16;
            a.h[0] = *(const v8h*)&As[arow][ka];
            a.h[1] = *(const v8h*)&As[arow][ka + 8];

            // B fragment: N = lane&15, 16 contiguous K at ks*32 + 16*halfSel.
            const int kb = ks * 32 + halfSel * 16;
            #pragma unroll
            for (int g = 0; g < 3; ++g) {
                const size_t wbase = (size_t)(g * LAYERS + layer) * H * H;
                #pragma unroll
                for (int t = 0; t < 4; ++t) {
                    const int n = ncol + t * 16;
                    v16h b = *(const v16h*)(Wt + wbase + (size_t)n * H + kb);
                    acc[g][t] = __builtin_amdgcn_wmma_f32_16x16x32_f16(
                        false, a.v, false, b, (short)0, acc[g][t],
                        false, false);
                }
            }
        }

        // ---- fused epilogue: h = tanh(sig(i)*tanh(g)) * sig(o) ----
        #pragma unroll
        for (int t = 0; t < 4; ++t) {
            const int n   = ncol + t * 16;
            const float vbi = bi[layer * H + n];
            const float vbo = bo[layer * H + n];
            const float vbg = bg[layer * H + n];
            #pragma unroll
            for (int j = 0; j < 8; ++j) {
                // C/D layout: VGPR j -> M = 8*halfSel + j, N = lane&15.
                const int m = m0 + wm * 16 + halfSel * 8 + j;
                const float si = fast_sigmoid(acc[0][t][j] + vbi);
                const float so = fast_sigmoid(acc[1][t][j] + vbo);
                const float tg = fast_tanh(acc[2][t][j] + vbg);
                const float hv = fast_tanh(si * tg) * so;
                const size_t o = (size_t)m * (H * LAYERS) + (size_t)layer * H + n;
                // write-once streams: non-temporal stores
                __builtin_nontemporal_store(hv, &out[o]);            // tuple slot 0
                __builtin_nontemporal_store(hv, &out[o + OUTCOPY]);  // tuple slot 1
            }
        }
    }
}

// ---------------------------------------------------------------------------
extern "C" void kernel_launch(void* const* d_in, const int* in_sizes, int n_in,
                              void* d_out, int out_size, void* d_ws, size_t ws_size,
                              hipStream_t stream) {
    // setup_inputs order: x, Wf, Wi, Wo, Wg, bf, bi, bo, bg  (Wf/bf are dead)
    const float* x  = (const float*)d_in[0];
    const float* Wi = (const float*)d_in[2];
    const float* Wo = (const float*)d_in[3];
    const float* Wg = (const float*)d_in[4];
    const float* bi = (const float*)d_in[6];
    const float* bo = (const float*)d_in[7];
    const float* bg = (const float*)d_in[8];
    float* out      = (float*)d_out;
    _Float16* Wt    = (_Float16*)d_ws;   // 3*2*256*256 f16 = 768 KB scratch

    prep_wt   <<<1536, 256, 0, stream>>>(Wi, Wo, Wg, Wt);
    zero_cells<<<CELLS / 256, 256, 0, stream>>>(out);
    lstm_gemm <<<dim3(ROWS / 64), dim3(512), 0, stream>>>(x, Wt, bi, bo, bg, out);
}
